// LLaDA2MoeSparseMoeBlock_27161373179909
// MI455X (gfx1250) — matrix operand order repaired and checked by reading, not compiled, not run on hardware
//
#include <hip/hip_runtime.h>
#include <hip/hip_bf16.h>

// ---------------------------------------------------------------------------
// LLaDA2 MoE sparse block for MI455X (gfx1250, wave32, WMMA bf16)
//   T=2048 tokens, H=2048, I=1024, E=16 experts, top-4, group-limited routing
// Pipeline:
//   init -> cvt x to bf16 -> route -> prefix scan -> per-expert token lists
//   -> stage1 (silu(x G^T) * (x U^T) -> bf16 M) for shared + experts
//   -> stage2 (M D^T) : shared writes out, experts atomically scatter-add
// GEMM cores: double-buffered LDS, A tiles streamed with
// global_load_async_to_lds_b128 (ASYNCcnt), B tiles register-staged with
// native fp32->bf16 converts, 8x v_wmma_f32_16x16x32_bf16 per K-step.
// ---------------------------------------------------------------------------

#define T_TOK 2048
#define H_DIM 2048
#define I_DIM 1024
#define N_EXP 16
#define TOPK  4

typedef __attribute__((ext_vector_type(16))) __bf16          v16bf;
typedef __attribute__((ext_vector_type(4)))  __bf16          v4bf;
typedef __attribute__((ext_vector_type(8)))  float           v8f;
typedef __attribute__((ext_vector_type(16))) unsigned short  v16us;
typedef __attribute__((ext_vector_type(8)))  unsigned short  v8us;
typedef __attribute__((ext_vector_type(4)))  unsigned short  v4us;

// Native fp32 -> bf16 (RTNE) conversions; lower to hardware cvt ops.
__device__ __forceinline__ unsigned short f2bf(float f) {
  union { __bf16 b; unsigned short u; } t;
  t.b = (__bf16)f;
  return t.u;
}

__device__ __forceinline__ v4us cvt4(float4 v) {
  union { v4bf b; v4us u; } t;
  t.b[0] = (__bf16)v.x; t.b[1] = (__bf16)v.y;
  t.b[2] = (__bf16)v.z; t.b[3] = (__bf16)v.w;
  return t.u;
}

union FragAB { v16us u; v16bf b; };

// A fragment: 16x32 bf16 tile. lane 0-15: row M=lane, K = 0..7 & 16..23
//             lane 16-31: row M=lane-16, K = 8..15 & 24..31
__device__ __forceinline__ v16bf load_a_frag(const unsigned short* tile,
                                             int row0, int lane, int lda) {
  int r  = row0 + (lane & 15);
  int kb = (lane >> 4) ? 8 : 0;
  const unsigned short* p = tile + r * lda;
  v8us lo = *(const v8us*)(p + kb);
  v8us hi = *(const v8us*)(p + kb + 16);
  FragAB f;
#pragma unroll
  for (int i = 0; i < 8; ++i) { f.u[i] = lo[i]; f.u[i + 8] = hi[i]; }
  return f.b;
}

// B fragment: 32x16 bf16. lane 0-15: col N=lane, K=0..7 & 16..23; lane 16-31: K=8..15 & 24..31
__device__ __forceinline__ v16bf load_b_frag(const unsigned short* tile,
                                             int col0, int lane, int ldb) {
  int c  = col0 + (lane & 15);
  int kb = (lane >> 4) ? 8 : 0;
  const unsigned short* p = tile + c * ldb + kb;
  v8us lo = *(const v8us*)(p);
  v8us hi = *(const v8us*)(p + 16);
  FragAB f;
#pragma unroll
  for (int i = 0; i < 8; ++i) { f.u[i] = lo[i]; f.u[i + 8] = hi[i]; }
  return f.b;
}

__device__ __forceinline__ v8f wmma_bf16(v16bf a, v16bf b, v8f c) {
  return __builtin_amdgcn_wmma_f32_16x16x32_bf16(false, a, false, b,
                                                 (short)0, c, false, false);
}

// CDNA5 async VMEM -> LDS copy, 16 bytes per lane, tracked with ASYNCcnt.
__device__ __forceinline__ void async_copy_b128(unsigned lds_byte_off,
                                                const void* gaddr) {
  asm volatile("global_load_async_to_lds_b128 %0, %1, off"
               :: "v"(lds_byte_off), "v"(gaddr) : "memory");
}

__device__ __forceinline__ void wait_async0() {
#if __has_builtin(__builtin_amdgcn_s_wait_asynccnt)
  __builtin_amdgcn_s_wait_asynccnt(0);
#else
  asm volatile("s_wait_asynccnt 0x0" ::: "memory");
#endif
}

__device__ __forceinline__ unsigned lds_off_u32(const void* p) {
  // Generic pointers to LDS carry the wave-relative LDS offset in the low
  // 32 bits.
  return (unsigned)(size_t)p;
}

// ---------------------------------------------------------------------------
// Small kernels: init / convert / routing / scan / list building
// ---------------------------------------------------------------------------

__global__ void init_kernel(int* cnt) {
  if (threadIdx.x < N_EXP) cnt[threadIdx.x] = 0;
}

__global__ void cvt_x_kernel(const float* __restrict__ x,
                             unsigned short* __restrict__ xb, int n4) {
  int i = blockIdx.x * blockDim.x + threadIdx.x;
  if (i < n4) {
    float4 v = *(const float4*)(x + (size_t)i * 4);
    *(v4us*)(xb + (size_t)i * 4) = cvt4(v);
  }
}

__global__ __launch_bounds__(128)
void route_kernel(const float* __restrict__ x, const float* __restrict__ gw,
                  const float* __restrict__ bias, int* __restrict__ sel,
                  float* __restrict__ wsel, int* __restrict__ cnt) {
  int t = blockIdx.x, tid = threadIdx.x;
  __shared__ float part[8][16];
  __shared__ float logits[16];
  int e = tid & 15, chunk = tid >> 4;                 // 16 experts x 8 chunks
  const float* xr = x + (size_t)t * H_DIM;
  const float* g  = gw + (size_t)e * H_DIM;
  float acc = 0.f;
  int h0 = chunk * (H_DIM / 8);
  for (int h = h0; h < h0 + H_DIM / 8; ++h) acc += xr[h] * g[h];
  part[chunk][e] = acc;
  __syncthreads();
  if (tid < 16) {
    float s = 0.f;
#pragma unroll
    for (int c = 0; c < 8; ++c) s += part[c][tid];
    logits[tid] = s;
  }
  __syncthreads();
  if (tid == 0) {
    float sc[16], bi[16];
#pragma unroll
    for (int i = 0; i < 16; ++i) {
      float s = 1.f / (1.f + __expf(-logits[i]));
      sc[i] = s; bi[i] = s + bias[i];
    }
    float gs[4];
#pragma unroll
    for (int gi = 0; gi < 4; ++gi) {                  // top-2 sum per group
      float m1 = -1e30f, m2 = -1e30f;
#pragma unroll
      for (int j = 0; j < 4; ++j) {
        float v = bi[gi * 4 + j];
        if (v > m1) { m2 = m1; m1 = v; } else if (v > m2) m2 = v;
      }
      gs[gi] = m1 + m2;
    }
    int g1 = 0;
    for (int gi = 1; gi < 4; ++gi) if (gs[gi] > gs[g1]) g1 = gi;
    int g2 = -1;
    for (int gi = 0; gi < 4; ++gi) {
      if (gi == g1) continue;
      if (g2 < 0 || gs[gi] > gs[g2]) g2 = gi;
    }
    bool used[16];
#pragma unroll
    for (int i = 0; i < 16; ++i) used[i] = false;
    int sk[4]; float wk[4]; float wsum = 0.f;
#pragma unroll
    for (int k = 0; k < TOPK; ++k) {                  // top-4 among 2 groups
      int best = -1;
      for (int i = 0; i < 16; ++i) {
        int gi = i >> 2;
        if (gi != g1 && gi != g2) continue;
        if (used[i]) continue;
        if (best < 0 || bi[i] > bi[best]) best = i;
      }
      used[best] = true; sk[k] = best; wk[k] = sc[best]; wsum += sc[best];
    }
    float inv = 2.5f / (wsum + 1e-20f);               // norm_topk * scaling
#pragma unroll
    for (int k = 0; k < TOPK; ++k) {
      sel[t * TOPK + k]  = sk[k];
      wsel[t * TOPK + k] = wk[k] * inv;
      atomicAdd(&cnt[sk[k]], 1);
    }
  }
}

__global__ void scan_kernel(const int* __restrict__ cnt, int* __restrict__ offs) {
  if (threadIdx.x == 0) {
    int s = 0;
    for (int e = 0; e < N_EXP; ++e) { offs[e] = s; s += cnt[e]; }
  }
}

// Deterministic token-ascending per-expert lists via ballot prefix.
__global__ void build_lists_kernel(const int* __restrict__ sel,
                                   const float* __restrict__ wsel,
                                   const int* __restrict__ offs,
                                   int* __restrict__ rowidx,
                                   float* __restrict__ roww) {
  int e = blockIdx.x, lane = threadIdx.x;              // 16 blocks x 32 lanes
  int cursor = offs[e];
  for (int base = 0; base < T_TOK; base += 32) {
    int t = base + lane;
    int slot = -1;
#pragma unroll
    for (int k = 0; k < TOPK; ++k) if (sel[t * TOPK + k] == e) slot = k;
    unsigned int m32 = (unsigned int)__ballot(slot >= 0);
    int pre = __popc(m32 & ((lane == 0) ? 0u : ((1u << lane) - 1u)));
    if (slot >= 0) {
      rowidx[cursor + pre] = t;
      roww[cursor + pre]   = wsel[t * TOPK + slot];
    }
    cursor += __popc(m32);
  }
}

// ---------------------------------------------------------------------------
// Stage 1: M = silu(X G^T) * (X U^T)   (128 tokens x 64 i, K over H)
// Double-buffered; A async-to-LDS; B register-staged fp32->bf16.
// ---------------------------------------------------------------------------
__global__ __launch_bounds__(256)
void stage1_kernel(const unsigned short* __restrict__ Xb,
                   const int* __restrict__ rowidx, const int* __restrict__ offs,
                   const int* __restrict__ cnts,
                   const float* __restrict__ Gw, const float* __restrict__ Uw,
                   unsigned short* __restrict__ Mout) {
  const int LDA = 40, LDB = 40;
  const int NITER = H_DIM / 32;
  __shared__ unsigned short As[2][128 * 40];
  __shared__ unsigned short Bg[2][64 * 40];
  __shared__ unsigned short Bu[2][64 * 40];

  int e = blockIdx.z;
  const float* G = Gw; const float* U = Uw;
  int rbase = 0, nrows = T_TOK;
  if (rowidx) {
    G = Gw + (size_t)e * I_DIM * H_DIM;
    U = Uw + (size_t)e * I_DIM * H_DIM;
    rbase = offs[e]; nrows = cnts[e];
  }
  int r0 = blockIdx.y * 128;
  if (r0 >= nrows) return;
  int n0 = blockIdx.x * 64;

  int tid = threadIdx.x, lane = tid & 31, w = tid >> 5;
  int rowgrp = w >> 1, colgrp = w & 1;                 // 4x2 wave layout

  // ---- loop-invariant addressing: A gather (2 x 16B chunks / thread) ----
  int arow0 = tid >> 2,         akc0 = tid & 3;
  int arow1 = (tid + 256) >> 2, akc1 = (tid + 256) & 3;
  int gr0 = r0 + arow0, gr1 = r0 + arow1;
  int tok0 = rowidx ? ((gr0 < nrows) ? rowidx[rbase + gr0] : 0) : gr0;
  int tok1 = rowidx ? ((gr1 < nrows) ? rowidx[rbase + gr1] : 0) : gr1;
  const unsigned short* ap0 = Xb + (size_t)tok0 * H_DIM + akc0 * 8;
  const unsigned short* ap1 = Xb + (size_t)tok1 * H_DIM + akc1 * 8;
  unsigned asb0 = (unsigned)(arow0 * LDA + akc0 * 8) * 2u;   // bytes
  unsigned asb1 = (unsigned)(arow1 * LDA + akc1 * 8) * 2u;
  unsigned as_base0 = lds_off_u32(&As[0][0]);
  unsigned as_base1 = lds_off_u32(&As[1][0]);

  // ---- B addressing: 2 float4 chunks / thread / matrix ----
  int brow0 = tid >> 3,         bc0 = tid & 7;
  int brow1 = (tid + 256) >> 3, bc1 = (tid + 256) & 7;
  const float* gp0 = G + (size_t)(n0 + brow0) * H_DIM + bc0 * 4;
  const float* gp1 = G + (size_t)(n0 + brow1) * H_DIM + bc1 * 4;
  const float* uq0 = U + (size_t)(n0 + brow0) * H_DIM + bc0 * 4;
  const float* uq1 = U + (size_t)(n0 + brow1) * H_DIM + bc1 * 4;
  int bl0 = brow0 * LDB + bc0 * 4;                     // ushort index
  int bl1 = brow1 * LDB + bc1 * 4;

  v8f accG[2][2], accU[2][2];
#pragma unroll
  for (int i = 0; i < 2; ++i)
#pragma unroll
    for (int j = 0; j < 2; ++j)
#pragma unroll
      for (int q = 0; q < 8; ++q) { accG[i][j][q] = 0.f; accU[i][j][q] = 0.f; }

  float4 rg0, rg1, ru0, ru1;

  auto issueA = [&](int buf, int k0) {
    unsigned base = buf ? as_base1 : as_base0;
    async_copy_b128(base + asb0, (const void*)(ap0 + k0));
    async_copy_b128(base + asb1, (const void*)(ap1 + k0));
  };
  auto loadB = [&](int k0) {
    rg0 = *(const float4*)(gp0 + k0);
    rg1 = *(const float4*)(gp1 + k0);
    ru0 = *(const float4*)(uq0 + k0);
    ru1 = *(const float4*)(uq1 + k0);
  };
  auto storeB = [&](int buf) {
    *(v4us*)(&Bg[buf][bl0]) = cvt4(rg0);
    *(v4us*)(&Bg[buf][bl1]) = cvt4(rg1);
    *(v4us*)(&Bu[buf][bl0]) = cvt4(ru0);
    *(v4us*)(&Bu[buf][bl1]) = cvt4(ru1);
  };

  // Prologue: tile 0 into buffer 0.
  issueA(0, 0);
  loadB(0);
  storeB(0);
  wait_async0();
  __syncthreads();

  for (int iter = 0; iter < NITER; ++iter) {
    int cur = iter & 1, nxt = cur ^ 1;
    int kn = (iter + 1) * 32;
    bool more = (iter + 1) < NITER;
    if (more) { issueA(nxt, kn); loadB(kn); }          // overlap with compute

    v16bf a0  = load_a_frag(&As[cur][0], rowgrp * 32,      lane, LDA);
    v16bf a1  = load_a_frag(&As[cur][0], rowgrp * 32 + 16, lane, LDA);
    v16bf bg0 = load_b_frag(&Bg[cur][0], colgrp * 32,      lane, LDB);
    v16bf bg1 = load_b_frag(&Bg[cur][0], colgrp * 32 + 16, lane, LDB);
    v16bf bu0 = load_b_frag(&Bu[cur][0], colgrp * 32,      lane, LDB);
    v16bf bu1 = load_b_frag(&Bu[cur][0], colgrp * 32 + 16, lane, LDB);

    accG[0][0] = wmma_bf16(a0, bg0, accG[0][0]);
    accG[0][1] = wmma_bf16(a0, bg1, accG[0][1]);
    accG[1][0] = wmma_bf16(a1, bg0, accG[1][0]);
    accG[1][1] = wmma_bf16(a1, bg1, accG[1][1]);
    accU[0][0] = wmma_bf16(a0, bu0, accU[0][0]);
    accU[0][1] = wmma_bf16(a0, bu1, accU[0][1]);
    accU[1][0] = wmma_bf16(a1, bu0, accU[1][0]);
    accU[1][1] = wmma_bf16(a1, bu1, accU[1][1]);

    if (more) storeB(nxt);
    wait_async0();
    __syncthreads();
  }

  // Epilogue: silu(g)*u -> bf16 M
  int halo = (lane >> 4) * 8;
#pragma unroll
  for (int i = 0; i < 2; ++i)
#pragma unroll
    for (int j = 0; j < 2; ++j)
#pragma unroll
      for (int v = 0; v < 8; ++v) {
        int m  = rowgrp * 32 + i * 16 + halo + v;
        int gr = r0 + m;
        if (gr < nrows) {
          int col = n0 + colgrp * 32 + j * 16 + (lane & 15);
          float gg = accG[i][j][v], uu = accU[i][j][v];
          float s = gg / (1.f + __expf(-gg));          // silu
          Mout[(size_t)(rbase + gr) * I_DIM + col] = f2bf(s * uu);
        }
      }
}

// ---------------------------------------------------------------------------
// Stage 2: Y = M D^T  (128 rows x 128 h, K over I)
// Double-buffered; A async-to-LDS; B register-staged fp32->bf16.
// ---------------------------------------------------------------------------
__global__ __launch_bounds__(256)
void stage2_kernel(const unsigned short* __restrict__ Mi,
                   const int* __restrict__ rowidx, const int* __restrict__ offs,
                   const int* __restrict__ cnts, const float* __restrict__ roww,
                   const float* __restrict__ Dw, float* __restrict__ out) {
  const int LDA = 40, LDB = 40;
  const int NITER = I_DIM / 32;
  __shared__ unsigned short As[2][128 * 40];
  __shared__ unsigned short Bd[2][128 * 40];

  int e = blockIdx.z;
  const float* D = Dw;
  int rbase = 0, nrows = T_TOK;
  if (rowidx) {
    D = Dw + (size_t)e * H_DIM * I_DIM;
    rbase = offs[e]; nrows = cnts[e];
  }
  int r0 = blockIdx.y * 128;
  if (r0 >= nrows) return;
  int n0 = blockIdx.x * 128;

  int tid = threadIdx.x, lane = tid & 31, w = tid >> 5;
  int rowgrp = w >> 1, colgrp = w & 1;

  // ---- A addressing (rows are contiguous compacted M rows) ----
  int arow0 = tid >> 2,         akc0 = tid & 3;
  int arow1 = (tid + 256) >> 2, akc1 = (tid + 256) & 3;
  int gr0 = r0 + arow0; if (gr0 >= nrows) gr0 = nrows - 1;
  int gr1 = r0 + arow1; if (gr1 >= nrows) gr1 = nrows - 1;
  const unsigned short* ap0 = Mi + (size_t)(rbase + gr0) * I_DIM + akc0 * 8;
  const unsigned short* ap1 = Mi + (size_t)(rbase + gr1) * I_DIM + akc1 * 8;
  unsigned asb0 = (unsigned)(arow0 * LDA + akc0 * 8) * 2u;
  unsigned asb1 = (unsigned)(arow1 * LDA + akc1 * 8) * 2u;
  unsigned as_base0 = lds_off_u32(&As[0][0]);
  unsigned as_base1 = lds_off_u32(&As[1][0]);

  // ---- B addressing: 4 float4 chunks / thread ----
  int brow[4], bcc[4];
  const float* bp[4];
  int bl[4];
#pragma unroll
  for (int c = 0; c < 4; ++c) {
    int idx = tid + c * 256;
    brow[c] = idx >> 3; bcc[c] = idx & 7;
    bp[c] = D + (size_t)(n0 + brow[c]) * I_DIM + bcc[c] * 4;
    bl[c] = brow[c] * LDB + bcc[c] * 4;
  }

  v8f acc[2][4];
#pragma unroll
  for (int i = 0; i < 2; ++i)
#pragma unroll
    for (int j = 0; j < 4; ++j)
#pragma unroll
      for (int q = 0; q < 8; ++q) acc[i][j][q] = 0.f;

  float4 rb[4];

  auto issueA = [&](int buf, int k0) {
    unsigned base = buf ? as_base1 : as_base0;
    async_copy_b128(base + asb0, (const void*)(ap0 + k0));
    async_copy_b128(base + asb1, (const void*)(ap1 + k0));
  };
  auto loadB = [&](int k0) {
#pragma unroll
    for (int c = 0; c < 4; ++c) rb[c] = *(const float4*)(bp[c] + k0);
  };
  auto storeB = [&](int buf) {
#pragma unroll
    for (int c = 0; c < 4; ++c) *(v4us*)(&Bd[buf][bl[c]]) = cvt4(rb[c]);
  };

  issueA(0, 0);
  loadB(0);
  storeB(0);
  wait_async0();
  __syncthreads();

  for (int iter = 0; iter < NITER; ++iter) {
    int cur = iter & 1, nxt = cur ^ 1;
    int kn = (iter + 1) * 32;
    bool more = (iter + 1) < NITER;
    if (more) { issueA(nxt, kn); loadB(kn); }

    v16bf a0 = load_a_frag(&As[cur][0], rowgrp * 32,      lane, LDA);
    v16bf a1 = load_a_frag(&As[cur][0], rowgrp * 32 + 16, lane, LDA);
#pragma unroll
    for (int j = 0; j < 4; ++j) {
      v16bf b = load_b_frag(&Bd[cur][0], colgrp * 64 + j * 16, lane, LDB);
      acc[0][j] = wmma_bf16(a0, b, acc[0][j]);
      acc[1][j] = wmma_bf16(a1, b, acc[1][j]);
    }

    if (more) storeB(nxt);
    wait_async0();
    __syncthreads();
  }

  int halo = (lane >> 4) * 8;
#pragma unroll
  for (int i = 0; i < 2; ++i)
#pragma unroll
    for (int j = 0; j < 4; ++j)
#pragma unroll
      for (int v = 0; v < 8; ++v) {
        int m  = rowgrp * 32 + i * 16 + halo + v;
        int gr = r0 + m;
        if (gr < nrows) {
          int col = n0 + colgrp * 64 + j * 16 + (lane & 15);
          float val = acc[i][j][v];
          if (rowidx) {
            int   tok = rowidx[rbase + gr];
            float wt  = roww[rbase + gr];
            atomicAdd(out + (size_t)tok * H_DIM + col, wt * val);
          } else {
            out[(size_t)gr * H_DIM + col] = val;
          }
        }
      }
}

// ---------------------------------------------------------------------------
// Host launch
// ---------------------------------------------------------------------------
extern "C" void kernel_launch(void* const* d_in, const int* in_sizes, int n_in,
                              void* d_out, int out_size, void* d_ws, size_t ws_size,
                              hipStream_t stream) {
  const float* x  = (const float*)d_in[0];
  const float* gw = (const float*)d_in[1];
  const float* eb = (const float*)d_in[2];
  const float* gp = (const float*)d_in[3];
  const float* up = (const float*)d_in[4];
  const float* dp = (const float*)d_in[5];
  const float* sg = (const float*)d_in[6];
  const float* su = (const float*)d_in[7];
  const float* sd = (const float*)d_in[8];
  float* out = (float*)d_out;

  char* ws = (char*)d_ws;
  size_t o = 0;
  unsigned short* Xb = (unsigned short*)(ws + o); o += (size_t)T_TOK * H_DIM * 2;        // 8 MB
  unsigned short* Ms = (unsigned short*)(ws + o); o += (size_t)T_TOK * I_DIM * 2;        // 4 MB
  unsigned short* Me = (unsigned short*)(ws + o); o += (size_t)T_TOK * TOPK * I_DIM * 2; // 16 MB
  int*   sel    = (int*)  (ws + o); o += (size_t)T_TOK * TOPK * 4;
  float* wsel   = (float*)(ws + o); o += (size_t)T_TOK * TOPK * 4;
  int*   rowidx = (int*)  (ws + o); o += (size_t)T_TOK * TOPK * 4;
  float* roww   = (float*)(ws + o); o += (size_t)T_TOK * TOPK * 4;
  int*   cnt    = (int*)  (ws + o); o += 64;
  int*   offs   = (int*)  (ws + o); o += 64;

  init_kernel<<<1, 32, 0, stream>>>(cnt);

  int n4 = T_TOK * H_DIM / 4;
  cvt_x_kernel<<<(n4 + 255) / 256, 256, 0, stream>>>(x, Xb, n4);

  route_kernel<<<T_TOK, 128, 0, stream>>>(x, gw, eb, sel, wsel, cnt);
  scan_kernel<<<1, 1, 0, stream>>>(cnt, offs);
  build_lists_kernel<<<N_EXP, 32, 0, stream>>>(sel, wsel, offs, rowidx, roww);

  // Shared expert stage 1 + expert stage 1
  stage1_kernel<<<dim3(I_DIM / 64, T_TOK / 128, 1), 256, 0, stream>>>(
      Xb, nullptr, nullptr, nullptr, sg, su, Ms);
  stage1_kernel<<<dim3(I_DIM / 64, T_TOK / 128, N_EXP), 256, 0, stream>>>(
      Xb, rowidx, offs, cnt, gp, up, Me);

  // Shared stage 2 writes out (base), then experts scatter-add on top.
  stage2_kernel<<<dim3(H_DIM / 128, T_TOK / 128, 1), 256, 0, stream>>>(
      Ms, nullptr, nullptr, nullptr, nullptr, sd, out);
  stage2_kernel<<<dim3(H_DIM / 128, T_TOK / 128, N_EXP), 256, 0, stream>>>(
      Me, rowidx, offs, cnt, roww, dp, out);
}
